// DLRM_Net_67499706023937
// MI455X (gfx1250) — compile-verified, matrix operand order checked by reading, not run on hardware
//
#include <hip/hip_runtime.h>
#include <math.h>

#define BATCH    4096
#define T_TABLES 26
#define L_POOL   20
#define N_TAB    100000
#define M_SPA    64
#define NI       27              // 1 dense + 26 sparse (real rows)
#define NIP      32              // padded rows per batch in Tcat
#define TCAT_LD  (NIP * M_SPA)   // 2048 bf16 per batch row
#define R_DIM    415
#define R_PAD    416             // padded K for top MLP (multiple of 32)

typedef __bf16 bf16_t;
typedef bf16_t v16bf __attribute__((ext_vector_type(16)));
typedef bf16_t v8bf  __attribute__((ext_vector_type(8)));
typedef bf16_t v2bf  __attribute__((ext_vector_type(2)));
typedef float  v8f   __attribute__((ext_vector_type(8)));

// ---------------------------------------------------------------------------
// Branch-free bf16 fragment load for V_WMMA_F32_16X16X32_BF16 (ISA 7.12.2):
// lane L holds row (L&15); lanes<16 take K = k0+{0..7,16..23}, lanes>=16 take
// K = k0+{8..15,24..31}. p = row base (row stride = Np, multiple of 32 bf16 ->
// 64B aligned rows; khalf offset is 16B aligned) -> two global_load_b128.
// Same layout serves A (row=M of X) and B (row=N of W) for X @ W^T.
// ---------------------------------------------------------------------------
__device__ __forceinline__ v16bf load_frag(const bf16_t* __restrict__ p,
                                           int k0, int khalf) {
  v8bf lo = *(const v8bf*)(p + k0 + khalf);
  v8bf hi = *(const v8bf*)(p + k0 + 16 + khalf);
  v16bf f;
#pragma unroll
  for (int e = 0; e < 8; ++e) { f[e] = lo[e]; f[8 + e] = hi[e]; }
  return f;
}

// ---------------------------------------------------------------------------
// f32 [M x N] -> zero-padded bf16 [Mp x Np]
// ---------------------------------------------------------------------------
__global__ void cvt_pad(const float* __restrict__ src, bf16_t* __restrict__ dst,
                        int M, int N, int Mp, int Np) {
  int e = blockIdx.x * blockDim.x + threadIdx.x;
  if (e >= Mp * Np) return;
  int r = e / Np, c = e % Np;
  float v = (r < M && c < N) ? src[(size_t)r * N + c] : 0.0f;
  dst[e] = (bf16_t)v;
}

// Zero Tcat pad rows 27..31 for every batch (rows contiguous per batch).
__global__ void zero_tcat_pad(bf16_t* __restrict__ Tcat) {
  int e = blockIdx.x * blockDim.x + threadIdx.x;
  const int per_b = (NIP - NI) * M_SPA;              // 320
  if (e >= BATCH * per_b) return;
  int b = e / per_b, rem = e % per_b;
  Tcat[(size_t)b * TCAT_LD + NI * M_SPA + rem] = (bf16_t)0.0f;
}

// ---------------------------------------------------------------------------
// Out[i,j] = act( sum_k A[i,k]*W[j,k] + bias[j] ),  A,W bf16 row-major with
// padded row stride Np (multiple of 32). Each wave: one 16x(16*JT) tile —
// one A fragment reused across JT W fragments -> JT WMMAs per K-chunk.
// ACT: 1=relu, 2=sigmoid.  F32OUT: store f32 (else bf16).
// ---------------------------------------------------------------------------
template<int ACT, int JT, bool F32OUT>
__global__ void gemm_bf16(const bf16_t* __restrict__ A,
                          const bf16_t* __restrict__ W,
                          const float* __restrict__ bias,
                          void* __restrict__ OutV,
                          int Mreal, int Np, int ldo, int mtg, int tiles) {
  const int lane = threadIdx.x & 31;
  const int tile = blockIdx.x * (blockDim.x >> 5) + (threadIdx.x >> 5);
  if (tile >= tiles) return;                         // wave-uniform
  const int ti  = tile / mtg;                        // 16-row batch tile
  const int tjg = tile % mtg;                        // group of JT col tiles

  const int col   = lane & 15;
  const int khalf = (lane >> 4) * 8;
  const bf16_t* pA = A + (size_t)(ti * 16 + (lane & 15)) * Np;

  const bf16_t* pW[JT];
  v8f c[JT];
#pragma unroll
  for (int s = 0; s < JT; ++s) {
    int jt = tjg * JT + s;
    pW[s] = W + (size_t)(jt * 16 + (lane & 15)) * Np;
    int j = jt * 16 + col;
    float bv = (j < Mreal) ? bias[j] : 0.0f;
    v8f init = {bv, bv, bv, bv, bv, bv, bv, bv};
    c[s] = init;
  }

  for (int k0 = 0; k0 < Np; k0 += 32) {
    v16bf a = load_frag(pA, k0, khalf);
#pragma unroll
    for (int s = 0; s < JT; ++s) {
      v16bf w = load_frag(pW[s], k0, khalf);
      c[s] = __builtin_amdgcn_wmma_f32_16x16x32_bf16(false, a, false, w,
                                                     (short)0, c[s], false, false);
    }
  }

  const int rbase = (lane >> 4) * 8;                 // D: VGPR v -> row v+rbase
#pragma unroll
  for (int s = 0; s < JT; ++s) {
    int j = (tjg * JT + s) * 16 + col;
#pragma unroll
    for (int v = 0; v < 8; ++v) {
      int i = ti * 16 + rbase + v;
      float x = c[s][v];
      if (ACT == 1) x = fmaxf(x, 0.0f);
      if (ACT == 2) x = 1.0f / (1.0f + __expf(-x));
      if (j < Mreal) {
        if (F32OUT) ((float*)OutV)[(size_t)i * ldo + j] = x;
        else        ((bf16_t*)OutV)[(size_t)i * ldo + j] = (bf16_t)x;
      }
    }
  }
}

// ---------------------------------------------------------------------------
// EmbeddingBag sum-pool: one wave per (table, batch); each lane owns 2 floats
// of the 64-wide row (256B coalesced per lookup). f32 accumulate, bf16 store
// into Tcat[b][1+t][:].
// ---------------------------------------------------------------------------
__global__ void embed_pool(const int* __restrict__ indices,
                           const float* __restrict__ tables,
                           bf16_t* __restrict__ Tcat) {
  const int lane = threadIdx.x & 31;
  const int wid  = (blockIdx.x * blockDim.x + threadIdx.x) >> 5;
  if (wid >= T_TABLES * BATCH) return;               // wave-uniform
  const int t = wid / BATCH;
  const int b = wid % BATCH;

  const int*   ix  = indices + ((size_t)t * BATCH + b) * L_POOL;
  const float* tab = tables + (size_t)t * N_TAB * M_SPA;

  float2 acc = make_float2(0.0f, 0.0f);
#pragma unroll
  for (int l = 0; l < L_POOL; ++l) {
    int r = ix[l];                                   // wave-uniform -> SMEM
    float2 v = ((const float2*)(tab + (size_t)r * M_SPA))[lane];
    acc.x += v.x; acc.y += v.y;
  }
  v2bf o; o[0] = (bf16_t)acc.x; o[1] = (bf16_t)acc.y;
  ((v2bf*)(Tcat + (size_t)b * TCAT_LD + (1 + t) * M_SPA))[lane] = o;
}

// ---------------------------------------------------------------------------
// Interaction: per batch (one wave), Z = T T^T, T = Tcat[b] (32x64 bf16,
// rows 27..31 zero). A and B fragments are identical (Gram matrix) -> 2 loads,
// 3 WMMAs per K-chunk (the (0,1) tile never hits the strict lower triangle).
// Scatter tril into R[b][64 + i*(i-1)/2 + j] (bf16), copy dense into R[b][0:64],
// zero the pad column R[b][415].
// ---------------------------------------------------------------------------
__device__ __forceinline__ void tri_scatter(bf16_t* __restrict__ Rb, v8f z,
                                            int ioff, int joff, int lane) {
  const int col = lane & 15;
  const int rbase = (lane >> 4) * 8;
#pragma unroll
  for (int v = 0; v < 8; ++v) {
    int i = ioff + rbase + v;
    int j = joff + col;
    if (i < NI && j < i) Rb[M_SPA + (i * (i - 1)) / 2 + j] = (bf16_t)z[v];
  }
}

__global__ void interact(const bf16_t* __restrict__ Tcat,
                         bf16_t* __restrict__ R) {
  const int lane = threadIdx.x & 31;
  const int b = blockIdx.x * (blockDim.x >> 5) + (threadIdx.x >> 5);
  if (b >= BATCH) return;                            // wave-uniform

  const bf16_t* T = Tcat + (size_t)b * TCAT_LD;
  const int khalf = (lane >> 4) * 8;
  const bf16_t* p0 = T + (size_t)(lane & 15) * M_SPA;
  const bf16_t* p1 = T + (size_t)(16 + (lane & 15)) * M_SPA;

  v8f z00 = {0,0,0,0,0,0,0,0}, z10 = {0,0,0,0,0,0,0,0}, z11 = {0,0,0,0,0,0,0,0};
#pragma unroll
  for (int k0 = 0; k0 < M_SPA; k0 += 32) {
    v16bf f0 = load_frag(p0, k0, khalf);
    v16bf f1 = load_frag(p1, k0, khalf);
    z00 = __builtin_amdgcn_wmma_f32_16x16x32_bf16(false, f0, false, f0,
                                                  (short)0, z00, false, false);
    z10 = __builtin_amdgcn_wmma_f32_16x16x32_bf16(false, f1, false, f0,
                                                  (short)0, z10, false, false);
    z11 = __builtin_amdgcn_wmma_f32_16x16x32_bf16(false, f1, false, f1,
                                                  (short)0, z11, false, false);
  }

  bf16_t* Rb = R + (size_t)b * R_PAD;
  ((v2bf*)Rb)[lane] = ((const v2bf*)T)[lane];        // dense 64 bf16
  if (lane == 0) Rb[R_DIM] = (bf16_t)0.0f;           // pad column
  tri_scatter(Rb, z00,  0,  0, lane);
  tri_scatter(Rb, z10, 16,  0, lane);
  tri_scatter(Rb, z11, 16, 16, lane);
}

// ---------------------------------------------------------------------------
extern "C" void kernel_launch(void* const* d_in, const int* in_sizes, int n_in,
                              void* d_out, int out_size, void* d_ws, size_t ws_size,
                              hipStream_t stream) {
  const float* dense_x = (const float*)d_in[0];
  const int*   indices = (const int*)d_in[1];
  const float* tables  = (const float*)d_in[2];
  const float* w_bot0 = (const float*)d_in[3];  const float* b_bot0 = (const float*)d_in[4];
  const float* w_bot1 = (const float*)d_in[5];  const float* b_bot1 = (const float*)d_in[6];
  const float* w_bot2 = (const float*)d_in[7];  const float* b_bot2 = (const float*)d_in[8];
  const float* w_top0 = (const float*)d_in[9];  const float* b_top0 = (const float*)d_in[10];
  const float* w_top1 = (const float*)d_in[11]; const float* b_top1 = (const float*)d_in[12];
  const float* w_top2 = (const float*)d_in[13]; const float* b_top2 = (const float*)d_in[14];
  float* out = (float*)d_out;

  // --- workspace carve-up (bf16 buffers, 256B aligned offsets) ---
  char* ws = (char*)d_ws;
  size_t off = 0;
  auto carve = [&](size_t bytes) {
    void* p = ws + off;
    off += (bytes + 255) & ~(size_t)255;
    return p;
  };
  bf16_t* Xp   = (bf16_t*)carve((size_t)BATCH * 32 * 2);       // dense_x padded
  bf16_t* wb0  = (bf16_t*)carve((size_t)512 * 32 * 2);
  bf16_t* wb1  = (bf16_t*)carve((size_t)256 * 512 * 2);
  bf16_t* wb2  = (bf16_t*)carve((size_t)64 * 256 * 2);
  bf16_t* wt0  = (bf16_t*)carve((size_t)512 * R_PAD * 2);
  bf16_t* wt1  = (bf16_t*)carve((size_t)256 * 512 * 2);
  bf16_t* wt2  = (bf16_t*)carve((size_t)16 * 256 * 2);
  bf16_t* h1   = (bf16_t*)carve((size_t)BATCH * 512 * 2);
  bf16_t* h2   = (bf16_t*)carve((size_t)BATCH * 256 * 2);
  bf16_t* Tcat = (bf16_t*)carve((size_t)BATCH * TCAT_LD * 2);
  bf16_t* R    = (bf16_t*)carve((size_t)BATCH * R_PAD * 2);
  bf16_t* t1   = (bf16_t*)carve((size_t)BATCH * 512 * 2);
  bf16_t* t2   = (bf16_t*)carve((size_t)BATCH * 256 * 2);

  auto pad = [&](const float* s, bf16_t* d, int M, int N, int Mp, int Np) {
    int tot = Mp * Np;
    cvt_pad<<<(tot + 255) / 256, 256, 0, stream>>>(s, d, M, N, Mp, Np);
  };

  // Stage 0: convert/pad inputs & weights to bf16.
  pad(dense_x, Xp,  BATCH, 13,  BATCH, 32);
  pad(w_bot0,  wb0, 512,   13,  512,   32);
  pad(w_bot1,  wb1, 256,  512,  256,  512);
  pad(w_bot2,  wb2, 64,   256,  64,   256);
  pad(w_top0,  wt0, 512, R_DIM, 512, R_PAD);
  pad(w_top1,  wt1, 256,  512,  256,  512);
  pad(w_top2,  wt2, 1,    256,  16,   256);
  {
    int tot = BATCH * (NIP - NI) * M_SPA;
    zero_tcat_pad<<<(tot + 255) / 256, 256, 0, stream>>>(Tcat);
  }

  const int WPB = 8;                                 // waves per block
  auto gemm4 = [&](const bf16_t* A, const bf16_t* W, const float* bias,
                   bf16_t* Out, int M, int Np, int ldo) {
    int mtg = M / 64;                                // JT=4 col-tile groups
    int tiles = (BATCH / 16) * mtg;
    gemm_bf16<1, 4, false><<<(tiles + WPB - 1) / WPB, WPB * 32, 0, stream>>>(
        A, W, bias, Out, M, Np, ldo, mtg, tiles);
  };

  // Bottom MLP: 13->512->256->64 (ReLU); last layer lands in Tcat row 0.
  gemm4(Xp, wb0, b_bot0, h1, 512, 32, 512);
  gemm4(h1, wb1, b_bot1, h2, 256, 512, 256);
  gemm4(h2, wb2, b_bot2, Tcat, 64, 256, TCAT_LD);

  // Embedding bag -> Tcat rows 1..26.
  {
    int waves = T_TABLES * BATCH;
    embed_pool<<<(waves + 7) / 8, 256, 0, stream>>>(indices, tables, Tcat);
  }

  // Interaction -> R [4096 x 416] bf16.
  interact<<<(BATCH + 7) / 8, 256, 0, stream>>>(Tcat, R);

  // Top MLP: 416->512->256 (ReLU) then 256->1 (sigmoid, f32 to d_out).
  gemm4(R,  wt0, b_top0, t1, 512, R_PAD, 512);
  gemm4(t1, wt1, b_top1, t2, 256, 512, 256);
  {
    int tiles = BATCH / 16;                          // mtg = 1 (Mpad=16, JT=1)
    gemm_bf16<2, 1, true><<<(tiles + WPB - 1) / WPB, WPB * 32, 0, stream>>>(
        t2, wt2, b_top2, out, 1, 256, 1, 1, tiles);
  }
}
